// LSTM_91036126806921
// MI455X (gfx1250) — matrix operand order repaired and checked by reading, not compiled, not run on hardware
//
#include <hip/hip_runtime.h>
#include <hip/hip_bf16.h>

// ---------------------------------------------------------------------------
// Persistent fused LSTM for MI455X (gfx1250, wave32, WMMA + TDM).
//   a_t = [x_t | h_{t-1}] * [Wx;Wh] + b   via v_wmma_f32_16x16x32_bf16
// 32 blocks, each owns 16 h-columns (=> 64 gate columns). B tile cached in
// LDS for the entire kernel. h_{t-1} staged into LDS by the Tensor Data
// Mover (tensor_load_to_lds + s_wait_tensorcnt). Grid-wide atomic barrier
// between timesteps.
// ---------------------------------------------------------------------------

typedef __attribute__((ext_vector_type(16))) __bf16 v16bf;
typedef __attribute__((ext_vector_type(8)))  __bf16 v8bf;
typedef __attribute__((ext_vector_type(4)))  __bf16 v4bf;
typedef __attribute__((ext_vector_type(8)))  float  v8f;
typedef __attribute__((ext_vector_type(4)))  unsigned int v4u;
typedef __attribute__((ext_vector_type(8)))  int    v8i;
typedef __attribute__((ext_vector_type(4)))  int    v4i;

#ifndef __has_builtin
#define __has_builtin(x) 0
#endif
#if __has_builtin(__builtin_amdgcn_tensor_load_to_lds) && \
    __has_builtin(__builtin_amdgcn_s_wait_tensorcnt)
#define HAVE_TDM 1
#else
#define HAVE_TDM 0
#endif

#define N_      64
#define T_      1024
#define D_      512
#define H_      512
#define GCOLS   2048      // 4*H_
#define NBLK    32
#define THREADS 256

// workspace layout (bytes)
#define WS_W    0                       // W_bf16: 1024 x 2048 bf16 = 4 MB
#define WS_H0   (4u*1024u*1024u)        // h ping buffer: 64x512 bf16 = 64 KB
#define WS_H1   (WS_H0 + 64u*1024u)     // h pong buffer
#define WS_BAR  (WS_H1 + 64u*1024u)     // grid barrier counters

// LDS layout (dynamic):
//   x_t stage 64KB | h stage 64KB | B fragments 128KB | gate scratch 16KB
#define LDS_X_OFF 0
#define LDS_H_OFF (64*1024)
#define LDS_B_OFF (128*1024)
#define LDS_G_OFF (256*1024)
#define SMEM_BYTES (272*1024)

__device__ __forceinline__ float sigmoid_fast(float v) {
    return 1.0f / (1.0f + __expf(-v));
}
__device__ __forceinline__ float tanh_fast(float v) {
    float e = __expf(2.0f * v);
    return 1.0f - 2.0f / (e + 1.0f);
}

__device__ __forceinline__ void grid_barrier(unsigned* cnt, unsigned* gen, unsigned nblk) {
    __syncthreads();
    if (threadIdx.x == 0) {
        __threadfence();
        unsigned g = __hip_atomic_load(gen, __ATOMIC_RELAXED, __HIP_MEMORY_SCOPE_AGENT);
        unsigned prev = __hip_atomic_fetch_add(cnt, 1u, __ATOMIC_ACQ_REL, __HIP_MEMORY_SCOPE_AGENT);
        if (prev == nblk - 1u) {
            __hip_atomic_store(cnt, 0u, __ATOMIC_RELAXED, __HIP_MEMORY_SCOPE_AGENT);
            __hip_atomic_fetch_add(gen, 1u, __ATOMIC_RELEASE, __HIP_MEMORY_SCOPE_AGENT);
        } else {
            while (__hip_atomic_load(gen, __ATOMIC_ACQUIRE, __HIP_MEMORY_SCOPE_AGENT) == g) {
                __builtin_amdgcn_s_sleep(8);
            }
        }
        __threadfence();
    }
    __syncthreads();
}

// Build combined bf16 weight matrix  W = [Wx ; Wh]  (1024 x 2048, row-major)
__global__ void lstm_conv_w(const float* __restrict__ Wx, const float* __restrict__ Wh,
                            __bf16* __restrict__ Wb) {
    size_t i = (size_t)blockIdx.x * blockDim.x + threadIdx.x;
    if (i >= (size_t)1024 * GCOLS) return;
    int k = (int)(i / GCOLS);
    int f = (int)(i % GCOLS);
    float v = (k < D_) ? Wx[(size_t)k * GCOLS + f] : Wh[(size_t)(k - D_) * GCOLS + f];
    Wb[i] = (__bf16)v;
}

__global__ void lstm_conv_h0(const float* __restrict__ h0, __bf16* __restrict__ hb) {
    int i = blockIdx.x * blockDim.x + threadIdx.x;
    if (i < N_ * H_) hb[i] = (__bf16)h0[i];
}

__global__ __launch_bounds__(THREADS, 1)
void lstm_persistent(const float* __restrict__ x, const float* __restrict__ bias,
                     const __bf16* __restrict__ Wb,
                     __bf16* __restrict__ hbuf0, __bf16* __restrict__ hbuf1,
                     float* __restrict__ out, unsigned* __restrict__ bar) {
    extern __shared__ char smem[];
    __bf16* ldsX = (__bf16*)(smem + LDS_X_OFF);   // 64 x 512 bf16, row stride 512
    __bf16* ldsH = (__bf16*)(smem + LDS_H_OFF);   // 64 x 512 bf16, row stride 512
    __bf16* ldsB = (__bf16*)(smem + LDS_B_OFF);   // pre-swizzled B fragments
    float*  ldsG = (float*)(smem + LDS_G_OFF);    // 64 x 64 gate scratch

    const int tid  = threadIdx.x;
    const int lane = tid & 31;
    const int wave = tid >> 5;
    const int blk  = blockIdx.x;

    unsigned* bar_cnt = bar;
    unsigned* bar_gen = bar + 16;

    // ---- Stage this block's B tile (K=1024 x 64 gate cols) into LDS once,
    //      already in this code's B-fragment order (k contiguous per lane half):
    //      chunk(nt, ks, khalf, col)[kk] = W[ks*32 + khalf*16 + kk][gcol]
    for (int idx = tid; idx < 1024 * 64; idx += THREADS) {
        int k = idx >> 6;                 // 0..1023
        int c = idx & 63;                 // local gate col
        int gcol = (c >> 4) * H_ + blk * 16 + (c & 15);
        __bf16 v = Wb[(size_t)k * GCOLS + gcol];
        int ks    = k >> 5;
        int khalf = (k >> 4) & 1;
        int kk    = k & 15;
        int dst = ((((c >> 4) * 32 + ks) * 2 + khalf) * 16 + (c & 15)) * 16 + kk;
        ldsB[dst] = v;
    }

    // ---- Per-thread elementwise ownership: column ec of this block's 16,
    //      4 rows (n) per thread.  c-state lives in registers (c0 == 0).
    const int   ec   = tid & 15;
    const int   hcol = blk * 16 + ec;
    float cst[4] = {0.f, 0.f, 0.f, 0.f};
    const float bi = bias[0 * H_ + hcol];
    const float bf = bias[1 * H_ + hcol];
    const float bo = bias[2 * H_ + hcol];
    const float bg = bias[3 * H_ + hcol];

    // ---- GEMM wave->tile mapping: 16 C tiles (4 mt x 4 nt), 2 per wave
    const int mt    = wave & 3;
    const int nt0   = (wave >> 2) * 2;
    const int arow  = mt * 16 + (lane & 15);
    const int khalf = lane >> 4;
    const int bcol  = lane & 15;

    __syncthreads();

    for (int t = 0; t < T_; ++t) {
        const __bf16* hrd = (t & 1) ? hbuf1 : hbuf0;
        __bf16*       hwr = (t & 1) ? hbuf0 : hbuf1;

        // ---- Stage x_t (cvt f32->bf16) into ldsX
        for (int idx = tid; idx < N_ * D_ / 4; idx += THREADS) {
            int n  = idx >> 7;            // 128 float4 per row
            int d4 = idx & 127;
            const float4 v = *(const float4*)(x + ((size_t)n * T_ + t) * D_ + d4 * 4);
            v4bf p = {(__bf16)v.x, (__bf16)v.y, (__bf16)v.z, (__bf16)v.w};
            *(v4bf*)(ldsX + n * 512 + d4 * 4) = p;
        }

        // ---- Stage h_{t-1} into ldsH via the Tensor Data Mover.
        //      64x512 bf16 tile, contiguous: tensor_dim0=tile_dim0=512,
        //      tensor_dim1=tile_dim1=64, stride 512, data_size=2B.
#if HAVE_TDM
        if (wave == 0) {
            unsigned long long ga = (unsigned long long)(uintptr_t)hrd;
            v4u g0;
            g0[0] = 1u;                                   // count=1, user mode
            g0[1] = (unsigned)LDS_H_OFF;                  // lds_addr (bytes)
            g0[2] = (unsigned)(ga & 0xFFFFFFFFu);         // global_addr[31:0]
            g0[3] = (unsigned)((ga >> 32) & 0x01FFFFFFu)  // global_addr[56:32]
                    | 0x80000000u;                        // type=2 ("image")
            v8i g1;
            g1[0] = (int)(1u << 16);      // workgroup_mask=0, data_size=1 (2B)
            g1[1] = (int)(512u << 16);    // tensor_dim0 = 512
            g1[2] = (int)(64u << 16);     // tensor_dim1 = 64
            g1[3] = (int)(512u << 16);    // tile_dim0 = 512
            g1[4] = 64;                   // tile_dim1 = 64 (tile_dim2 = 0)
            g1[5] = 512;                  // tensor_dim0_stride = 512
            g1[6] = 0;
            g1[7] = 0;
            v4i gz = {0, 0, 0, 0};
#if __clang_major__ >= 23
            v8i gz8 = {0, 0, 0, 0, 0, 0, 0, 0};
            __builtin_amdgcn_tensor_load_to_lds(g0, g1, gz, gz, gz8, 0);
#else
            __builtin_amdgcn_tensor_load_to_lds(g0, g1, gz, gz, 0);
#endif
            __builtin_amdgcn_s_wait_tensorcnt(0);
        }
#else
        for (int idx = tid; idx < N_ * H_ / 8; idx += THREADS) {
            int n = idx >> 6;             // 64 8-elem chunks per row
            int q = idx & 63;
            *(v8bf*)(ldsH + n * 512 + q * 8) = *(const v8bf*)(hrd + n * H_ + q * 8);
        }
#endif

        // gfx1250 prefetch of next timestep's x rows (global_prefetch_b8)
        if (t + 1 < T_) {
            int pn = tid & 63;
            int pq = tid >> 6;            // 4 quadrants of the 2KB row
            __builtin_prefetch(x + ((size_t)pn * T_ + (t + 1)) * D_ + pq * 128, 0, 1);
        }
        __syncthreads();

        // ---- K-loop: 32 x (K=32) bf16 WMMA, fp32 accumulate.
        //      First 16 k-steps read the x half, last 16 read the h half.
        v8f acc0 = {};
        v8f acc1 = {};
        const __bf16* aX = ldsX + arow * 512 + khalf * 16;
        const __bf16* aH = ldsH + arow * 512 + khalf * 16;

#define GEMM_STEP(APTR, KSG)                                                        \
        {                                                                           \
            v8bf alo = *(const v8bf*)(APTR);                                        \
            v8bf ahi = *(const v8bf*)((APTR) + 8);                                  \
            v16bf afrag = __builtin_shufflevector(alo, ahi,                         \
                0, 1, 2, 3, 4, 5, 6, 7, 8, 9, 10, 11, 12, 13, 14, 15);              \
            const v16bf bfrag0 = *(const v16bf*)(ldsB +                             \
                ((((nt0 + 0) * 32 + (KSG)) * 2 + khalf) * 16 + bcol) * 16);         \
            const v16bf bfrag1 = *(const v16bf*)(ldsB +                             \
                ((((nt0 + 1) * 32 + (KSG)) * 2 + khalf) * 16 + bcol) * 16);         \
            acc0 = __builtin_amdgcn_wmma_f32_16x16x32_bf16(                         \
                false, afrag, false, bfrag0, (short)0, acc0, false, false);         \
            acc1 = __builtin_amdgcn_wmma_f32_16x16x32_bf16(                         \
                false, afrag, false, bfrag1, (short)0, acc1, false, false);         \
        }

        #pragma unroll 4
        for (int ks = 0; ks < 16; ++ks) {
            GEMM_STEP(aX + ks * 32, ks)
        }
        #pragma unroll 4
        for (int ks = 0; ks < 16; ++ks) {
            GEMM_STEP(aH + ks * 32, ks + 16)
        }
#undef GEMM_STEP

        // ---- Spill C tiles to gate scratch (C layout: row = v + 8*(lane/16))
        #pragma unroll
        for (int v = 0; v < 8; ++v) {
            int row = mt * 16 + v + 8 * khalf;
            ldsG[row * 64 + (nt0 + 0) * 16 + bcol] = acc0[v];
            ldsG[row * 64 + (nt0 + 1) * 16 + bcol] = acc1[v];
        }
        __syncthreads();

        // ---- Elementwise LSTM cell update for 4 (n, hcol) pairs per thread
        #pragma unroll
        for (int j = 0; j < 4; ++j) {
            int n = (tid >> 4) + 16 * j;
            float ai = ldsG[n * 64 +  0 + ec] + bi;
            float af = ldsG[n * 64 + 16 + ec] + bf;
            float ao = ldsG[n * 64 + 32 + ec] + bo;
            float ag = ldsG[n * 64 + 48 + ec] + bg;
            float ig = sigmoid_fast(ai);
            float fg = sigmoid_fast(af);
            float og = sigmoid_fast(ao);
            float gg = tanh_fast(ag);
            float cn = fg * cst[j] + ig * gg;
            cst[j] = cn;
            float hn = og * tanh_fast(cn);
            out[((size_t)n * T_ + t) * H_ + hcol] = hn;
            hwr[n * H_ + hcol] = (__bf16)hn;
        }

        // ---- Publish h_t, wait for all blocks before next step
        grid_barrier(bar_cnt, bar_gen, NBLK);
    }
}

extern "C" void kernel_launch(void* const* d_in, const int* in_sizes, int n_in,
                              void* d_out, int out_size, void* d_ws, size_t ws_size,
                              hipStream_t stream) {
    const float* x  = (const float*)d_in[0];
    const float* h0 = (const float*)d_in[1];
    const float* Wx = (const float*)d_in[2];
    const float* Wh = (const float*)d_in[3];
    const float* b  = (const float*)d_in[4];
    float* out = (float*)d_out;

    char* ws = (char*)d_ws;
    __bf16*   Wb  = (__bf16*)(ws + WS_W);
    __bf16*   h0b = (__bf16*)(ws + WS_H0);
    __bf16*   h1b = (__bf16*)(ws + WS_H1);
    unsigned* bar = (unsigned*)(ws + WS_BAR);

    (void)in_sizes; (void)n_in; (void)out_size; (void)ws_size;

    // large dynamic LDS (272 KB of the 320 KB WGP pool); idempotent, called
    // unconditionally every launch (no static state in kernel_launch)
    hipFuncSetAttribute((const void*)lstm_persistent,
                        hipFuncAttributeMaxDynamicSharedMemorySize, SMEM_BYTES);

    // deterministic: barrier counters zeroed every launch (graph-capturable)
    hipMemsetAsync(bar, 0, 256, stream);

    {
        size_t tot = (size_t)1024 * GCOLS;
        lstm_conv_w<<<(unsigned)((tot + 255) / 256), 256, 0, stream>>>(Wx, Wh, Wb);
    }
    lstm_conv_h0<<<(N_ * H_ + 255) / 256, 256, 0, stream>>>(h0, h0b);

    lstm_persistent<<<NBLK, THREADS, SMEM_BYTES, stream>>>(x, b, Wb, h0b, h1b, out, bar);
}
